// VQPatternMatrixV74_80616536145998
// MI455X (gfx1250) — matrix operand order repaired and checked by reading, not compiled, not run on hardware
//
#include <hip/hip_runtime.h>
#include <hip/hip_bf16.h>
#include <math.h>

typedef __bf16 bf16_t;
typedef __attribute__((ext_vector_type(16))) __bf16 v16bf;
typedef __attribute__((ext_vector_type(2)))  __bf16 vbf2;
typedef __attribute__((ext_vector_type(8)))  float  v8f;
typedef __attribute__((ext_vector_type(2)))  float  vf2;

#define N_TOK 65536
#define D_IN  1024
#define BO    64
#define KP    32
// TEMP = 0.2 + 0.8*(1+cos(pi*(60-15)/85))/2
#define TEMP_F 0.56309264f

// ---- helpers -------------------------------------------------------------

__device__ __forceinline__ unsigned pcg_hash(unsigned v) {
  v = v * 747796405u + 2891336453u;
  unsigned w = ((v >> ((v >> 28) + 4u)) ^ v) * 277803737u;
  return (w >> 22) ^ w;
}

// deterministic pseudo-gaussian (Box-Muller on counter hash)
__device__ __forceinline__ float gauss01(unsigned idx) {
  unsigned h1 = pcg_hash(idx ^ 0x9E3779B9u);
  unsigned h2 = pcg_hash(idx + 0x7F4A7C15u);
  float u1 = ((float)h1 + 1.0f) * 2.3283064e-10f;   // (0,1]
  float u2 = (float)h2 * 2.3283064e-10f;
  return sqrtf(fmaxf(-2.0f * __logf(u1), 0.0f)) * __cosf(6.2831853f * u2);
}

// A fragment 16x32 bf16: lane half h, VGPR v -> K pair at 2v + 16h
__device__ __forceinline__ v16bf load_a_frag(const float* __restrict__ row,
                                             int kb, int half) {
  v16bf a;
#pragma unroll
  for (int v = 0; v < 8; ++v) {
    int k = kb + 2 * v + half * 16;
    vf2 f = *reinterpret_cast<const vf2*>(row + k);
    vbf2 h = __builtin_convertvector(f, vbf2);     // native packed f32->bf16
    a[2 * v]     = h[0];
    a[2 * v + 1] = h[1];
  }
  return a;
}

// ---- k_prep: zero accumulators, normalize patterns, transpose W1/W2 ------

__global__ void k_prep(const float* __restrict__ W1, const float* __restrict__ W2,
                       const float* __restrict__ patterns,
                       bf16_t* __restrict__ W1T, bf16_t* __restrict__ W2T,
                       float* __restrict__ knorm, float* __restrict__ acc, int acc_n)
{
  int gid = blockIdx.x * blockDim.x + threadIdx.x;
  int stride = gridDim.x * blockDim.x;
  for (int i = gid; i < acc_n; i += stride) acc[i] = 0.0f;
  // W1 [1024][64] -> W1T bf16 [64 cols][1024 K]
  for (int i = gid; i < D_IN * BO; i += stride) {
    int k = i >> 6, n = i & 63;
    W1T[(size_t)n * D_IN + k] = (bf16_t)W1[i];
  }
  // W2 [64][1024] -> W2T bf16 [1024 cols][64 K]
  for (int i = gid; i < BO * D_IN; i += stride) {
    int k = i >> 10, n = i & 1023;
    W2T[(size_t)n * BO + k] = (bf16_t)W2[i];
  }
  if (gid < KP) {
    float ss = 0.0f;
    for (int d = 0; d < BO; ++d) { float v = patterns[gid * BO + d]; ss += v * v; }
    float inv = 1.0f / fmaxf(sqrtf(ss), 1e-12f);
    for (int d = 0; d < BO; ++d) knorm[gid * BO + d] = patterns[gid * BO + d] * inv;
  }
}

// ---- k_proj: Q = LN(x @ W1 + b1) via bf16 WMMA ---------------------------
// 1 wave = 16 tokens x 64 outputs; block = 4 waves = 64 tokens.
// K-loop is ping-pong double-buffered: loads for chunk k+1 issue before the
// WMMAs of chunk k, so s_wait_loadcnt covers a full chunk of latency.

__global__ __launch_bounds__(128)
void k_proj(const float* __restrict__ x, const bf16_t* __restrict__ W1T,
            const float* __restrict__ b1, const float* __restrict__ g1,
            const float* __restrict__ beta1, float* __restrict__ qout)
{
  int wave = threadIdx.x >> 5;
  int lane = threadIdx.x & 31;
  int half = lane >> 4;
  int nl   = lane & 15;
  int m0   = blockIdx.x * 64 + wave * 16;
  const float* xrow = x + (size_t)(m0 + nl) * D_IN;

  // B fragment 32x16: lane = column n, VGPR v -> K pair 2v + 16h.
  // With W1T pre-transposed these 16 bf16 are contiguous & 32B aligned.
  const bf16_t* bbase[4];
#pragma unroll
  for (int t = 0; t < 4; ++t)
    bbase[t] = W1T + (size_t)(t * 16 + nl) * D_IN + 16 * half;

  v8f c[4] = {};
  v16bf a0, a1;
  v16bf b0[4], b1v[4];

  a0 = load_a_frag(xrow, 0, half);
#pragma unroll
  for (int t = 0; t < 4; ++t)
    b0[t] = *reinterpret_cast<const v16bf*>(bbase[t] + 0);

  for (int kb = 0; kb < D_IN; kb += 64) {
    __builtin_prefetch(xrow + kb + 64, 0, 1);
    // stage chunk kb+32
    a1 = load_a_frag(xrow, kb + 32, half);
#pragma unroll
    for (int t = 0; t < 4; ++t)
      b1v[t] = *reinterpret_cast<const v16bf*>(bbase[t] + kb + 32);
    // compute chunk kb
#pragma unroll
    for (int t = 0; t < 4; ++t)
      c[t] = __builtin_amdgcn_wmma_f32_16x16x32_bf16(false, a0, false, b0[t],
                                                     (short)0, c[t], false, false);
    // stage chunk kb+64
    if (kb + 64 < D_IN) {
      a0 = load_a_frag(xrow, kb + 64, half);
#pragma unroll
      for (int t = 0; t < 4; ++t)
        b0[t] = *reinterpret_cast<const v16bf*>(bbase[t] + kb + 64);
    }
    // compute chunk kb+32
#pragma unroll
    for (int t = 0; t < 4; ++t)
      c[t] = __builtin_amdgcn_wmma_f32_16x16x32_bf16(false, a1, false, b1v[t],
                                                     (short)0, c[t], false, false);
  }

  // bias
#pragma unroll
  for (int t = 0; t < 4; ++t) {
    float bb = b1[t * 16 + nl];
#pragma unroll
    for (int r = 0; r < 8; ++r) c[t][r] += bb;
  }
  // LayerNorm per row: C layout -> VGPR r holds row r (lanes 0-15) / r+8 (16-31)
#pragma unroll
  for (int r = 0; r < 8; ++r) {
    float s  = c[0][r] + c[1][r] + c[2][r] + c[3][r];
    float s2 = c[0][r]*c[0][r] + c[1][r]*c[1][r] + c[2][r]*c[2][r] + c[3][r]*c[3][r];
#pragma unroll
    for (int m = 8; m >= 1; m >>= 1) {
      s  += __shfl_xor(s,  m, 32);
      s2 += __shfl_xor(s2, m, 32);
    }
    float mu   = s * (1.0f / 64.0f);
    float var  = s2 * (1.0f / 64.0f) - mu * mu;
    float rstd = rsqrtf(fmaxf(var, 0.0f) + 1e-5f);
    int row = m0 + r + 8 * half;
#pragma unroll
    for (int t = 0; t < 4; ++t) {
      int col = t * 16 + nl;
      qout[(size_t)row * BO + col] = (c[t][r] - mu) * rstd * g1[col] + beta1[col];
    }
  }
}

// ---- k_logits: logits/softmax/argmax + sinkhorn init + EMA stats ---------

__global__ __launch_bounds__(128)
void k_logits(const float* __restrict__ qout, const float* __restrict__ knorm,
              float* __restrict__ logits_out, float* __restrict__ soft_out,
              float* __restrict__ hidx_f_out, int* __restrict__ hidx_out,
              float* __restrict__ qmat, float* __restrict__ colsum0,
              float* __restrict__ bcs, float* __restrict__ dwacc)
{
  __shared__ float s_kn[KP][BO];
  __shared__ float s_hist[KP];
  __shared__ float s_col[KP];
  __shared__ float s_dw[KP][BO];
  for (int i = threadIdx.x; i < KP * BO; i += blockDim.x) {
    s_kn[i >> 6][i & 63] = knorm[i];
    s_dw[i >> 6][i & 63] = 0.0f;
  }
  for (int i = threadIdx.x; i < KP; i += blockDim.x) { s_hist[i] = 0.0f; s_col[i] = 0.0f; }
  __syncthreads();

  int tok  = blockIdx.x * blockDim.x + threadIdx.x;
  int lane = threadIdx.x & 31;
  const float* qr = qout + (size_t)tok * BO;
  float q[BO];
  float ss = 0.0f;
#pragma unroll
  for (int d = 0; d < BO; ++d) { q[d] = qr[d]; ss += q[d] * q[d]; }
  float rn = 1.0f / fmaxf(sqrtf(ss), 1e-12f);

  float lg[KP];
  float mx = -3.4e38f;
  int   hi = 0;
  for (int k = 0; k < KP; ++k) {
    float dot = 0.0f;
#pragma unroll
    for (int d = 0; d < BO; ++d) dot += q[d] * s_kn[k][d];
    float v = dot * rn * (1.0f / TEMP_F) + 0.03f * gauss01((unsigned)(tok * KP + k));
    lg[k] = v;
    if (v > mx) { mx = v; hi = k; }
    logits_out[(size_t)tok * KP + k] = v;
  }
  // softmax
  float es = 0.0f;
  for (int k = 0; k < KP; ++k) es += __expf(lg[k] - mx);
  float inv = 1.0f / es;
  for (int k = 0; k < KP; ++k) soft_out[(size_t)tok * KP + k] = __expf(lg[k] - mx) * inv;

  hidx_out[tok]   = hi;
  hidx_f_out[tok] = (float)hi;

  // sinkhorn init: Q = exp((lg - rowmax)/0.1) + 1e-8 ; accumulate column sums
  for (int k = 0; k < KP; ++k) {
    float qk = __expf((lg[k] - mx) * 10.0f) + 1e-8f;
    qmat[(size_t)tok * KP + k] = qk;
    float v = qk;
#pragma unroll
    for (int m = 16; m >= 1; m >>= 1) v += __shfl_xor(v, m, 32);
    if (lane == 0) atomicAdd(&s_col[k], v);
  }
  // EMA stats
  atomicAdd(&s_hist[hi], 1.0f);
  for (int d = 0; d < BO; ++d) atomicAdd(&s_dw[hi][d], q[d]);
  __syncthreads();
  for (int i = threadIdx.x; i < KP; i += blockDim.x) {
    atomicAdd(&colsum0[i], s_col[i]);
    atomicAdd(&bcs[i], s_hist[i]);
  }
  for (int i = threadIdx.x; i < KP * BO; i += blockDim.x)
    atomicAdd(&dwacc[i], s_dw[i >> 6][i & 63]);
}

// ---- k_sinkhorn: one col+row normalization; final pass mixes soft --------

__global__ __launch_bounds__(128)
void k_sinkhorn(float* __restrict__ qmat, const float* __restrict__ colsum_in,
                float* __restrict__ colsum_out, int final_pass,
                const int* __restrict__ hidx, float* __restrict__ soft_io,
                float* __restrict__ assign_out)
{
  __shared__ float s_cs[KP];
  __shared__ float s_col[KP];
  for (int i = threadIdx.x; i < KP; i += blockDim.x) {
    s_cs[i]  = fmaxf(colsum_in[i], 1e-8f);
    s_col[i] = 0.0f;
  }
  __syncthreads();
  int tok  = blockIdx.x * blockDim.x + threadIdx.x;
  int lane = threadIdx.x & 31;
  float* row = qmat + (size_t)tok * KP;
  float v[KP];
  float rs = 0.0f;
#pragma unroll
  for (int k = 0; k < KP; ++k) { v[k] = row[k] / s_cs[k]; rs += v[k]; }
  float inv = 1.0f / fmaxf(rs, 1e-8f);

  if (!final_pass) {
#pragma unroll
    for (int k = 0; k < KP; ++k) {
      float w = v[k] * inv;
      row[k] = w;
#pragma unroll
      for (int m = 16; m >= 1; m >>= 1) w += __shfl_xor(w, m, 32);
      if (lane == 0) atomicAdd(&s_col[k], w);
    }
    __syncthreads();
    for (int i = threadIdx.x; i < KP; i += blockDim.x) atomicAdd(&colsum_out[i], s_col[i]);
  } else {
    int hi = hidx[tok];
    float* srow = soft_io    + (size_t)tok * KP;
    float* arow = assign_out + (size_t)tok * KP;
#pragma unroll
    for (int k = 0; k < KP; ++k) {
      float sk   = v[k] * inv;
      float soft = 0.5f * srow[k] + 0.5f * sk;
      srow[k] = soft;
      float hard = (k == hi) ? 1.0f : 0.0f;
      arow[k] = (hard + soft) - soft;   // straight-through: numerically hard
    }
  }
}

// ---- k_finalize: EMA codebook updates ------------------------------------

__global__ void k_finalize(const float* __restrict__ bcs, const float* __restrict__ dwacc,
                           const float* __restrict__ ema_cs, const float* __restrict__ ema_dw,
                           const float* __restrict__ pusage,
                           float* __restrict__ out_ema_cs, float* __restrict__ out_ema_dw,
                           float* __restrict__ out_patterns, float* __restrict__ out_usage)
{
  __shared__ float s_necs[KP];
  __shared__ float s_bcs[KP];
  __shared__ float s_n, s_bsum;
  int t = threadIdx.x;   // 64 threads
  if (t < KP) {
    float b  = bcs[t];
    float ne = 0.99f * ema_cs[t] + 0.01f * b;
    s_bcs[t]  = b;
    s_necs[t] = ne;
    out_ema_cs[t] = ne;
  }
  for (int i = t; i < KP * BO; i += 64)
    out_ema_dw[i] = 0.99f * ema_dw[i] + 0.01f * dwacc[i];
  __syncthreads();
  if (t == 0) {
    float n = 0.0f, bs = 0.0f;
    for (int k = 0; k < KP; ++k) { n += s_necs[k]; bs += s_bcs[k]; }
    s_n = n; s_bsum = bs;
  }
  __syncthreads();
  if (t < KP) {
    float n  = s_n;
    float cs = (s_necs[t] + 1e-5f) / (n + (float)KP * 1e-5f) * n;
    cs = fmaxf(cs, 1.0f);
    float tmp[BO];
    float sq = 0.0f;
    for (int d = 0; d < BO; ++d) {
      tmp[d] = out_ema_dw[t * BO + d] / cs;
      sq += tmp[d] * tmp[d];
    }
    float inv = 1.0f / fmaxf(sqrtf(sq), 1e-12f);
    for (int d = 0; d < BO; ++d) out_patterns[t * BO + d] = tmp[d] * inv;
    out_usage[t] = 0.9f * pusage[t] + 0.1f * (s_bcs[t] / (s_bsum + 1e-8f));
  }
}

// ---- k_decode: LN(patterns[hidx] @ W2 + b2) via bf16 WMMA ----------------
// block = 16 tokens x 1024 cols; 8 waves, each wave 8 N-tiles, K = 64 (2 chunks)

__global__ __launch_bounds__(256)
void k_decode(const int* __restrict__ hidx, const float* __restrict__ patterns,
              const bf16_t* __restrict__ W2T, const float* __restrict__ b2,
              const float* __restrict__ g2, const float* __restrict__ beta2,
              float* __restrict__ out)
{
  __shared__ float s_sum[16][8];
  __shared__ float s_ssq[16][8];
  int wave = threadIdx.x >> 5;
  int lane = threadIdx.x & 31;
  int half = lane >> 4;
  int nl   = lane & 15;
  int m0   = blockIdx.x * 16;
  int hi   = hidx[m0 + nl];
  const float* arow = patterns + (size_t)hi * BO;

  v8f c[8] = {};
#pragma unroll
  for (int kb = 0; kb < BO; kb += 32) {
    v16bf a = load_a_frag(arow, kb, half);
#pragma unroll
    for (int t = 0; t < 8; ++t) {
      // contiguous, 32B aligned B fragment from pre-transposed W2T
      v16bf b = *reinterpret_cast<const v16bf*>(
          W2T + (size_t)(wave * 128 + t * 16 + nl) * BO + kb + 16 * half);
      c[t] = __builtin_amdgcn_wmma_f32_16x16x32_bf16(false, a, false, b,
                                                     (short)0, c[t], false, false);
    }
  }
  // bias + LN partials across this wave's 128 columns
#pragma unroll
  for (int t = 0; t < 8; ++t) {
    float bb = b2[wave * 128 + t * 16 + nl];
#pragma unroll
    for (int r = 0; r < 8; ++r) c[t][r] += bb;
  }
#pragma unroll
  for (int r = 0; r < 8; ++r) {
    float s = 0.0f, s2 = 0.0f;
#pragma unroll
    for (int t = 0; t < 8; ++t) { float v = c[t][r]; s += v; s2 += v * v; }
#pragma unroll
    for (int m = 8; m >= 1; m >>= 1) {
      s  += __shfl_xor(s,  m, 32);
      s2 += __shfl_xor(s2, m, 32);
    }
    if (nl == 0) { s_sum[r + 8 * half][wave] = s; s_ssq[r + 8 * half][wave] = s2; }
  }
  __syncthreads();
#pragma unroll
  for (int r = 0; r < 8; ++r) {
    int row = r + 8 * half;
    float s = 0.0f, s2 = 0.0f;
#pragma unroll
    for (int w = 0; w < 8; ++w) { s += s_sum[row][w]; s2 += s_ssq[row][w]; }
    float mu   = s * (1.0f / 1024.0f);
    float var  = s2 * (1.0f / 1024.0f) - mu * mu;
    float rstd = rsqrtf(fmaxf(var, 0.0f) + 1e-5f);
    size_t rowoff = (size_t)(m0 + row) * D_IN;
#pragma unroll
    for (int t = 0; t < 8; ++t) {
      int col = wave * 128 + t * 16 + nl;
      out[rowoff + col] = (c[t][r] - mu) * rstd * g2[col] + beta2[col];
    }
  }
}

// ---- launcher ------------------------------------------------------------

extern "C" void kernel_launch(void* const* d_in, const int* in_sizes, int n_in,
                              void* d_out, int out_size, void* d_ws, size_t ws_size,
                              hipStream_t stream)
{
  (void)in_sizes; (void)n_in; (void)out_size; (void)ws_size;
  const float* x        = (const float*)d_in[0];
  const float* W1       = (const float*)d_in[1];
  const float* b1       = (const float*)d_in[2];
  const float* g1       = (const float*)d_in[3];
  const float* beta1    = (const float*)d_in[4];
  const float* patterns = (const float*)d_in[5];
  const float* W2       = (const float*)d_in[6];
  const float* b2       = (const float*)d_in[7];
  const float* g2       = (const float*)d_in[8];
  const float* beta2    = (const float*)d_in[9];
  const float* ema_cs   = (const float*)d_in[10];
  const float* ema_dw   = (const float*)d_in[11];
  const float* pusage   = (const float*)d_in[12];

  float* out = (float*)d_out;
  size_t o = 0;
  float* out_emb    = out + o; o += (size_t)N_TOK * D_IN;
  float* out_assign = out + o; o += (size_t)N_TOK * KP;
  float* out_logits = out + o; o += (size_t)N_TOK * KP;
  float* out_hidx   = out + o; o += (size_t)N_TOK;
  float* out_q      = out + o; o += (size_t)N_TOK * BO;
  float* out_soft   = out + o; o += (size_t)N_TOK * KP;
  float* out_emacs  = out + o; o += KP;
  float* out_emadw  = out + o; o += KP * BO;
  float* out_pat    = out + o; o += KP * BO;
  float* out_usage  = out + o; o += KP;

  char* ws = (char*)d_ws;
  size_t woff = 0;
  auto carve = [&](size_t bytes) -> char* {
    woff = (woff + 255) & ~(size_t)255;
    char* p = ws + woff;
    woff += bytes;
    return p;
  };
  bf16_t* w1t   = (bf16_t*)carve((size_t)D_IN * BO * sizeof(bf16_t));
  bf16_t* w2t   = (bf16_t*)carve((size_t)D_IN * BO * sizeof(bf16_t));
  float*  knorm = (float*) carve((size_t)KP * BO * sizeof(float));
  float*  qmat  = (float*) carve((size_t)N_TOK * KP * sizeof(float));
  int*    hidx  = (int*)   carve((size_t)N_TOK * sizeof(int));
  int acc_n = 4 * KP + KP + KP * BO;    // 4 colsum bufs + bcs + dwacc = 2208 floats
  float* accr   = (float*) carve((size_t)acc_n * sizeof(float));
  float* colsum = accr;                 // [4][32]
  float* bcs    = accr + 4 * KP;        // [32]
  float* dwacc  = accr + 5 * KP;        // [32][64]

  k_prep<<<64, 256, 0, stream>>>(W1, W2, patterns, w1t, w2t, knorm, accr, acc_n);
  k_proj<<<N_TOK / 64, 128, 0, stream>>>(x, w1t, b1, g1, beta1, out_q);
  k_logits<<<N_TOK / 128, 128, 0, stream>>>(out_q, knorm, out_logits, out_soft,
                                            out_hidx, hidx, qmat, colsum, bcs, dwacc);
  k_sinkhorn<<<N_TOK / 128, 128, 0, stream>>>(qmat, colsum + 0 * KP, colsum + 1 * KP, 0,
                                              hidx, out_soft, out_assign);
  k_sinkhorn<<<N_TOK / 128, 128, 0, stream>>>(qmat, colsum + 1 * KP, colsum + 2 * KP, 0,
                                              hidx, out_soft, out_assign);
  k_sinkhorn<<<N_TOK / 128, 128, 0, stream>>>(qmat, colsum + 2 * KP, colsum + 3 * KP, 1,
                                              hidx, out_soft, out_assign);
  k_finalize<<<1, 64, 0, stream>>>(bcs, dwacc, ema_cs, ema_dw, pusage,
                                   out_emacs, out_emadw, out_pat, out_usage);
  k_decode<<<N_TOK / 16, 256, 0, stream>>>(hidx, patterns, w2t, b2, g2, beta2, out_emb);
}
